// Tmix_77601469104911
// MI455X (gfx1250) — matrix operand order repaired and checked by reading, not compiled, not run on hardware
//
#include <hip/hip_runtime.h>
#include <hip/hip_bf16.h>
#include <stdint.h>
#include <stddef.h>

// ---- problem constants (match reference) ----
#define B_ 2
#define T_ 1024
#define C_ 1024
#define H_ 16
#define N_ 64
#define BT_ (B_ * T_)
#define BTC_ (BT_ * C_)
#define EPS_GN 6.4e-4f

typedef __bf16 bf16_t;
typedef __attribute__((ext_vector_type(16))) __bf16 v16bf;
typedef __attribute__((ext_vector_type(8)))  float  v8f;

__device__ __forceinline__ float sigm(float x) { return 1.0f / (1.0f + expf(-x)); }

__device__ __forceinline__ void wait_asynccnt0() {
  asm volatile("s_wait_asynccnt 0x0" ::: "memory");
}
// CDNA5 async DMA: global -> LDS, 16 bytes per lane, tracked by ASYNCcnt.
__device__ __forceinline__ void async_copy_b128(uint32_t lds_off, const void* gptr) {
  unsigned long long g = (unsigned long long)(uintptr_t)gptr;
  asm volatile("global_load_async_to_lds_b128 %0, %1, off"
               :: "v"(lds_off), "v"(g) : "memory");
}

// =====================================================================
// 1) time-shift + token mixes -> six bf16 activation buffers
// =====================================================================
__global__ __launch_bounds__(256) void mix_kernel(
    const float* __restrict__ x,
    const float* __restrict__ mr, const float* __restrict__ mw,
    const float* __restrict__ mk, const float* __restrict__ mv,
    const float* __restrict__ ma, const float* __restrict__ mg,
    bf16_t* __restrict__ xr, bf16_t* __restrict__ xw, bf16_t* __restrict__ xk,
    bf16_t* __restrict__ xv, bf16_t* __restrict__ xa, bf16_t* __restrict__ xg) {
  int idx = blockIdx.x * 256 + threadIdx.x;
  if (idx >= BTC_) return;
  int c  = idx & (C_ - 1);
  int bt = idx >> 10;           // C_ == 1024
  int t  = bt & (T_ - 1);
  float xv_ = x[idx];
  float prev = (t > 0) ? x[idx - C_] : 0.0f;
  float xx = prev - xv_;
  xr[idx] = (bf16_t)(xv_ + xx * mr[c]);
  xw[idx] = (bf16_t)(xv_ + xx * mw[c]);
  xk[idx] = (bf16_t)(xv_ + xx * mk[c]);
  xv[idx] = (bf16_t)(xv_ + xx * mv[c]);
  xa[idx] = (bf16_t)(xv_ + xx * ma[c]);
  xg[idx] = (bf16_t)(xv_ + xx * mg[c]);
}

// =====================================================================
// 2) conversion / small elementwise kernels
// =====================================================================
__global__ __launch_bounds__(256) void cvt_f32_bf16(const float* __restrict__ s,
                                                    bf16_t* __restrict__ d, int n) {
  int i = blockIdx.x * 256 + threadIdx.x;
  if (i < n) d[i] = (bf16_t)s[i];
}
__global__ __launch_bounds__(256) void tanh_cvt(const float* __restrict__ s,
                                                bf16_t* __restrict__ d, int n) {
  int i = blockIdx.x * 256 + threadIdx.x;
  if (i < n) d[i] = (bf16_t)tanhf(s[i]);
}
__global__ __launch_bounds__(256) void sig_cvt(const float* __restrict__ s,
                                               bf16_t* __restrict__ d, int n) {
  int i = blockIdx.x * 256 + threadIdx.x;
  if (i < n) d[i] = (bf16_t)sigm(s[i]);
}
// w = exp(-exp(w0 + wlin))  (in place on wlin)
__global__ __launch_bounds__(256) void decay_kernel(float* __restrict__ wl,
                                                    const float* __restrict__ w0) {
  int i = blockIdx.x * 256 + threadIdx.x;
  if (i < BTC_) wl[i] = expf(-expf(w0[i & (C_ - 1)] + wl[i]));
}
// a = sigmoid(a0 + alin)  (in place)
__global__ __launch_bounds__(256) void asig_kernel(float* __restrict__ al,
                                                   const float* __restrict__ a0) {
  int i = blockIdx.x * 256 + threadIdx.x;
  if (i < BTC_) al[i] = sigm(a0[i & (C_ - 1)] + al[i]);
}

// =====================================================================
// 3) generic bf16 WMMA GEMM:  C[M,N] = A[M,K] * B[K,N]   (fp32 accum)
//    M % 128 == 0, N % 64 == 0, K % 32 == 0 (true for all call sites)
//    block = 256 thr (8 waves), tile 128x64, K-step 32.
//    Double-buffered LDS; A tile staged via GLOBAL_LOAD_ASYNC_TO_LDS_B128
//    (ASYNCcnt) so the next tile's DMA overlaps this tile's WMMA chain.
// =====================================================================
#define GBM 128
#define GBN 64
#define GBK 32

__device__ __forceinline__ void stage_tile(
    bf16_t (*__restrict__ Asb)[GBK], bf16_t (*__restrict__ Bsb)[GBK + 2],
    const bf16_t* __restrict__ A, const bf16_t* __restrict__ Bm,
    int m0, int n0, int k0, int K, int N, int tid) {
  // ---- A tile 128x32: async DMA, 2 x 16B per thread ----
#pragma unroll
  for (int it = 0; it < 2; ++it) {
    int vi  = tid + it * 256;
    int row = vi >> 2;
    int col = (vi & 3) * 8;
    uint32_t lds = (uint32_t)(uintptr_t)&Asb[row][col];
    async_copy_b128(lds, A + (size_t)(m0 + row) * K + k0 + col);
  }
  // ---- B tile 32x64: regular 16B load + transpose scatter into Bs[n][k] ----
  {
    int row = tid >> 3;            // k (0..31)
    int col = (tid & 7) * 8;       // n
    bf16_t tmp[8];
    *(uint4*)tmp = *(const uint4*)(Bm + (size_t)(k0 + row) * N + n0 + col);
#pragma unroll
    for (int j = 0; j < 8; ++j) Bsb[col + j][row] = tmp[j];
  }
}

__global__ __launch_bounds__(256) void gemm_bf16_wmma(
    const bf16_t* __restrict__ A, const bf16_t* __restrict__ Bm,
    float* __restrict__ Cm, int M, int N, int K) {
  __shared__ __align__(16) bf16_t As[2][GBM][GBK];        // 2 x 8 KB
  __shared__ __align__(16) bf16_t Bs[2][GBN][GBK + 2];    // 2 x 4.25 KB

  const int tid  = threadIdx.x;
  const int wv   = tid >> 5;
  const int lane = tid & 31;
  const int half = lane >> 4;
  const int l16  = lane & 15;
  const int m0 = blockIdx.y * GBM;
  const int n0 = blockIdx.x * GBN;

  v8f acc[4];
#pragma unroll
  for (int c = 0; c < 4; ++c)
#pragma unroll
    for (int j = 0; j < 8; ++j) acc[c][j] = 0.0f;

  // prologue: stage tile 0 into buffer 0
  stage_tile(As[0], Bs[0], A, Bm, m0, n0, 0, K, N, tid);
  wait_asynccnt0();
  __syncthreads();

  int p = 0;
  for (int k0 = 0; k0 < K; k0 += GBK, p ^= 1) {
    // prefetch next tile into the other buffer (overlaps with WMMA below)
    if (k0 + GBK < K)
      stage_tile(As[p ^ 1], Bs[p ^ 1], A, Bm, m0, n0, k0 + GBK, K, N, tid);

    // --- A fragment (ISA 16-bit A 16x32 layout) ---
    v16bf af;
    const int arow = wv * 16 + l16;
#pragma unroll
    for (int i = 0; i < 8; ++i) {
      int kk2 = ((i & 4) ? 16 : 0) + half * 8 + (i & 3) * 2;
      af[2 * i]     = As[p][arow][kk2];
      af[2 * i + 1] = As[p][arow][kk2 + 1];
    }
    // --- preload all 4 B fragments, then issue WMMAs back-to-back ---
    v16bf bfr[4];
#pragma unroll
    for (int c = 0; c < 4; ++c) {
      const int bn = c * 16 + l16;
#pragma unroll
      for (int i = 0; i < 8; ++i) {
        int kk2 = ((i & 4) ? 16 : 0) + half * 8 + (i & 3) * 2;
        bfr[c][2 * i]     = Bs[p][bn][kk2];
        bfr[c][2 * i + 1] = Bs[p][bn][kk2 + 1];
      }
    }
#pragma unroll
    for (int c = 0; c < 4; ++c) {
      acc[c] = __builtin_amdgcn_wmma_f32_16x16x32_bf16(
          false, af, false, bfr[c], (short)0, acc[c], false, false);
    }

    wait_asynccnt0();   // own async DMAs for next tile are done
    __syncthreads();    // everyone's ds stores + reads of this tile are done
  }

  // --- store D (ISA 32-bit C/D 16x16 layout) ---
#pragma unroll
  for (int c = 0; c < 4; ++c) {
#pragma unroll
    for (int i = 0; i < 8; ++i) {
      int gm = m0 + wv * 16 + half * 8 + i;
      int gn = n0 + c * 16 + l16;
      Cm[(size_t)gm * N + gn] = acc[c][i];
    }
  }
}

// =====================================================================
// 4) per-head k normalization + k scaling (one wave per (bt,h) row)
//    kk = normalize(k*k_k) ; k = k*(1+(a-1)*k_a)
// =====================================================================
__global__ __launch_bounds__(32) void knorm_kernel(
    float* __restrict__ k, const float* __restrict__ a,
    const float* __restrict__ k_k, const float* __restrict__ k_a,
    float* __restrict__ kkout) {
  int row = blockIdx.x;              // bt*H + h
  int h = row & (H_ - 1);
  int bt = row >> 4;
  int lane = threadIdx.x;
  int base = bt * C_ + h * N_;
  int c0 = h * N_ + lane, c1 = c0 + 32;
  float k0v = k[base + lane], k1v = k[base + lane + 32];
  float q0 = k0v * k_k[c0], q1 = k1v * k_k[c1];
  float ss = q0 * q0 + q1 * q1;
#pragma unroll
  for (int m = 16; m >= 1; m >>= 1) ss += __shfl_xor(ss, m, 32);
  float inv = 1.0f / fmaxf(sqrtf(ss), 1e-12f);
  kkout[base + lane]      = q0 * inv;
  kkout[base + lane + 32] = q1 * inv;
  float a0v = a[base + lane], a1v = a[base + lane + 32];
  k[base + lane]      = k0v * (1.0f + (a0v - 1.0f) * k_a[c0]);
  k[base + lane + 32] = k1v * (1.0f + (a1v - 1.0f) * k_a[c1]);
}

// =====================================================================
// 5) WKV-7 sequential scan. One block per (b,h); 64 threads; thread i
//    owns state row S[i][:]. at = -kk, bt = kk*a computed inline.
// =====================================================================
__global__ __launch_bounds__(64) void wkv_kernel(
    const float* __restrict__ r, const float* __restrict__ w,
    const float* __restrict__ k, const float* __restrict__ v,
    const float* __restrict__ kk, const float* __restrict__ a,
    float* __restrict__ o) {
  int bh = blockIdx.x;
  int b = bh / H_, h = bh % H_;
  int i = threadIdx.x;               // state row 0..63
  float S[N_];
#pragma unroll
  for (int j = 0; j < N_; ++j) S[j] = 0.0f;

  __shared__ float rs[N_], ws[N_], ks[N_], avs[N_], bvs[N_];

  for (int t = 0; t < T_; ++t) {
    int base = (b * T_ + t) * C_ + h * N_;
    float kkv = kk[base + i];
    rs[i]  = r[base + i];
    ws[i]  = w[base + i];
    ks[i]  = k[base + i];
    avs[i] = -kkv;
    bvs[i] = kkv * a[base + i];
    float vi = v[base + i];
    __syncthreads();
    float sa = 0.0f;
#pragma unroll
    for (int j = 0; j < N_; ++j) sa += S[j] * avs[j];
    float out = 0.0f;
#pragma unroll
    for (int j = 0; j < N_; ++j) {
      S[j] = S[j] * ws[j] + sa * bvs[j] + vi * ks[j];
      out += S[j] * rs[j];
    }
    o[base + i] = out;
    __syncthreads();
  }
}

// =====================================================================
// 6) fused GroupNorm + rkv + rosa gating + gate-by-g, emits bf16 (o*g)
//    one block per (b,t); 256 threads x 4 channels
// =====================================================================
__global__ __launch_bounds__(256) void post_kernel(
    const float* __restrict__ o_in, const float* __restrict__ r,
    const float* __restrict__ k, const float* __restrict__ v,
    const float* __restrict__ g, const float* __restrict__ r_k,
    const float* __restrict__ ln_w, const float* __restrict__ ln_b,
    const int* __restrict__ ids, const float* __restrict__ emb,
    const float* __restrict__ rgate, bf16_t* __restrict__ og16) {
  __shared__ float red[256];
  int bt = blockIdx.x;
  int tid = threadIdx.x;
  int base = bt * C_;
  int c0 = tid * 4;                  // 4 channels, same head group (16 thr/head)

  float xv[4];
#pragma unroll
  for (int j = 0; j < 4; ++j) xv[j] = o_in[base + c0 + j];

  // group mean/var over 64 channels == 16 threads (lane-aligned halves)
  float s = xv[0] + xv[1] + xv[2] + xv[3];
#pragma unroll
  for (int m = 8; m >= 1; m >>= 1) s += __shfl_xor(s, m, 32);
  float mu = s * (1.0f / 64.0f);
  float vsum = 0.0f;
#pragma unroll
  for (int j = 0; j < 4; ++j) { float d = xv[j] - mu; vsum += d * d; }
#pragma unroll
  for (int m = 8; m >= 1; m >>= 1) vsum += __shfl_xor(vsum, m, 32);
  float inv = rsqrtf(vsum * (1.0f / 64.0f) + EPS_GN);
#pragma unroll
  for (int j = 0; j < 4; ++j) {
    int c = c0 + j;
    xv[j] = (xv[j] - mu) * inv * ln_w[c] + ln_b[c];
  }

  // rkv: per head sum of r*k*r_k, then += s*v
  int h = tid >> 4;
  float sr = 0.0f;
#pragma unroll
  for (int j = 0; j < 4; ++j) {
    int c = c0 + j;
    sr += r[base + c] * k[base + c] * r_k[h * N_ + (c & (N_ - 1))];
  }
#pragma unroll
  for (int m = 8; m >= 1; m >>= 1) sr += __shfl_xor(sr, m, 32);
#pragma unroll
  for (int j = 0; j < 4; ++j) xv[j] += sr * v[base + c0 + j];

  // rosa gate: mean over C of sigmoid(o * rosa_gate)
  float ls = 0.0f;
#pragma unroll
  for (int j = 0; j < 4; ++j) ls += sigm(xv[j] * rgate[c0 + j]);
  red[tid] = ls;
  __syncthreads();
  for (int off = 128; off > 0; off >>= 1) {
    if (tid < off) red[tid] += red[tid + off];
    __syncthreads();
  }
  float gate = red[0] * (1.0f / (float)C_);

  const float* erow = emb + (size_t)ids[bt] * C_;
#pragma unroll
  for (int j = 0; j < 4; ++j) {
    int c = c0 + j;
    float onew = xv[j] * (1.0f - gate) + erow[c] * gate;
    og16[base + c] = (bf16_t)(onew * g[base + c]);
  }
}

// =====================================================================
// host-side launcher
// =====================================================================
extern "C" void kernel_launch(void* const* d_in, const int* in_sizes, int n_in,
                              void* d_out, int out_size, void* d_ws, size_t ws_size,
                              hipStream_t stream) {
  const float* x       = (const float*)d_in[0];
  const int*   ids     = (const int*)d_in[1];
  const float* x_r     = (const float*)d_in[2];
  const float* x_w     = (const float*)d_in[3];
  const float* x_k     = (const float*)d_in[4];
  const float* x_v     = (const float*)d_in[5];
  const float* x_a     = (const float*)d_in[6];
  const float* x_g     = (const float*)d_in[7];
  const float* w0      = (const float*)d_in[8];
  const float* a0      = (const float*)d_in[9];
  // d_in[10] = v0, d_in[15] = v1, d_in[16] = v2 : v-blend is identity in fwd
  const float* w1      = (const float*)d_in[11];
  const float* w2      = (const float*)d_in[12];
  const float* a1      = (const float*)d_in[13];
  const float* a2      = (const float*)d_in[14];
  const float* g1      = (const float*)d_in[17];
  const float* g2      = (const float*)d_in[18];
  const float* k_k     = (const float*)d_in[19];
  const float* k_a     = (const float*)d_in[20];
  const float* r_k     = (const float*)d_in[21];
  const float* Wr      = (const float*)d_in[22];
  const float* Wk      = (const float*)d_in[23];
  const float* Wv      = (const float*)d_in[24];
  const float* Wo      = (const float*)d_in[25];
  const float* ln_w    = (const float*)d_in[26];
  const float* ln_b    = (const float*)d_in[27];
  const float* emb     = (const float*)d_in[28];
  const float* rgate   = (const float*)d_in[29];
  float* out = (float*)d_out;

  // ---- carve workspace ----
  char* p = (char*)d_ws;
  auto carve = [&](size_t bytes) -> void* {
    void* q = (void*)p;
    p += (bytes + 255) & ~(size_t)255;
    return q;
  };
  bf16_t* xr16 = (bf16_t*)carve((size_t)BTC_ * 2);
  bf16_t* xw16 = (bf16_t*)carve((size_t)BTC_ * 2);
  bf16_t* xk16 = (bf16_t*)carve((size_t)BTC_ * 2);
  bf16_t* xv16 = (bf16_t*)carve((size_t)BTC_ * 2);
  bf16_t* xa16 = (bf16_t*)carve((size_t)BTC_ * 2);
  bf16_t* xg16 = (bf16_t*)carve((size_t)BTC_ * 2);
  bf16_t* Wr16 = (bf16_t*)carve((size_t)C_ * C_ * 2);
  bf16_t* Wk16 = (bf16_t*)carve((size_t)C_ * C_ * 2);
  bf16_t* Wv16 = (bf16_t*)carve((size_t)C_ * C_ * 2);
  bf16_t* Wo16 = (bf16_t*)carve((size_t)C_ * C_ * 2);
  bf16_t* w1_16 = (bf16_t*)carve((size_t)C_ * 64 * 2);
  bf16_t* w2_16 = (bf16_t*)carve((size_t)64 * C_ * 2);
  bf16_t* a1_16 = (bf16_t*)carve((size_t)C_ * 64 * 2);
  bf16_t* a2_16 = (bf16_t*)carve((size_t)64 * C_ * 2);
  bf16_t* g1_16 = (bf16_t*)carve((size_t)C_ * 128 * 2);
  bf16_t* g2_16 = (bf16_t*)carve((size_t)128 * C_ * 2);
  bf16_t* hw16  = (bf16_t*)carve((size_t)BT_ * 64 * 2);
  bf16_t* ha16  = (bf16_t*)carve((size_t)BT_ * 64 * 2);
  bf16_t* hg16  = (bf16_t*)carve((size_t)BT_ * 128 * 2);
  bf16_t* og16  = (bf16_t*)carve((size_t)BTC_ * 2);
  float* rbuf = (float*)carve((size_t)BTC_ * 4);
  float* kbuf = (float*)carve((size_t)BTC_ * 4);
  float* vbuf = (float*)carve((size_t)BTC_ * 4);
  float* wlin = (float*)carve((size_t)BTC_ * 4);   // -> decay in place
  float* abuf = (float*)carve((size_t)BTC_ * 4);   // alin -> a in place
  float* gbuf = (float*)carve((size_t)BTC_ * 4);
  float* kkb  = (float*)carve((size_t)BTC_ * 4);
  float* obuf = (float*)carve((size_t)BTC_ * 4);
  float* hw   = (float*)carve((size_t)BT_ * 64 * 4);
  float* ha   = (float*)carve((size_t)BT_ * 64 * 4);
  float* hg   = (float*)carve((size_t)BT_ * 128 * 4);
  (void)ws_size; (void)n_in; (void)in_sizes; (void)out_size;

  const int EW = (BTC_ + 255) / 256;

  // 1) mixes
  mix_kernel<<<EW, 256, 0, stream>>>(x, x_r, x_w, x_k, x_v, x_a, x_g,
                                     xr16, xw16, xk16, xv16, xa16, xg16);
  // 2) weight conversions
  cvt_f32_bf16<<<(C_ * C_ + 255) / 256, 256, 0, stream>>>(Wr, Wr16, C_ * C_);
  cvt_f32_bf16<<<(C_ * C_ + 255) / 256, 256, 0, stream>>>(Wk, Wk16, C_ * C_);
  cvt_f32_bf16<<<(C_ * C_ + 255) / 256, 256, 0, stream>>>(Wv, Wv16, C_ * C_);
  cvt_f32_bf16<<<(C_ * C_ + 255) / 256, 256, 0, stream>>>(Wo, Wo16, C_ * C_);
  cvt_f32_bf16<<<(C_ * 64 + 255) / 256, 256, 0, stream>>>(w1, w1_16, C_ * 64);
  cvt_f32_bf16<<<(C_ * 64 + 255) / 256, 256, 0, stream>>>(w2, w2_16, 64 * C_);
  cvt_f32_bf16<<<(C_ * 64 + 255) / 256, 256, 0, stream>>>(a1, a1_16, C_ * 64);
  cvt_f32_bf16<<<(C_ * 64 + 255) / 256, 256, 0, stream>>>(a2, a2_16, 64 * C_);
  cvt_f32_bf16<<<(C_ * 128 + 255) / 256, 256, 0, stream>>>(g1, g1_16, C_ * 128);
  cvt_f32_bf16<<<(C_ * 128 + 255) / 256, 256, 0, stream>>>(g2, g2_16, 128 * C_);

  // 3) big projections r,k,v
  dim3 gBig(C_ / GBN, BT_ / GBM);
  gemm_bf16_wmma<<<gBig, 256, 0, stream>>>(xr16, Wr16, rbuf, BT_, C_, C_);
  gemm_bf16_wmma<<<gBig, 256, 0, stream>>>(xk16, Wk16, kbuf, BT_, C_, C_);
  gemm_bf16_wmma<<<gBig, 256, 0, stream>>>(xv16, Wv16, vbuf, BT_, C_, C_);

  // 4) w path: hw = xw@w1 (N=64) -> tanh -> @w2 -> decay
  gemm_bf16_wmma<<<dim3(1, BT_ / GBM), 256, 0, stream>>>(xw16, w1_16, hw, BT_, 64, C_);
  tanh_cvt<<<(BT_ * 64 + 255) / 256, 256, 0, stream>>>(hw, hw16, BT_ * 64);
  gemm_bf16_wmma<<<gBig, 256, 0, stream>>>(hw16, w2_16, wlin, BT_, C_, 64);
  decay_kernel<<<EW, 256, 0, stream>>>(wlin, w0);

  // 5) a path: ha = xa@a1 (N=64) -> @a2 -> sigmoid(a0+.)
  gemm_bf16_wmma<<<dim3(1, BT_ / GBM), 256, 0, stream>>>(xa16, a1_16, ha, BT_, 64, C_);
  cvt_f32_bf16<<<(BT_ * 64 + 255) / 256, 256, 0, stream>>>(ha, ha16, BT_ * 64);
  gemm_bf16_wmma<<<gBig, 256, 0, stream>>>(ha16, a2_16, abuf, BT_, C_, 64);
  asig_kernel<<<EW, 256, 0, stream>>>(abuf, a0);

  // 6) g path: hg = xg@g1 (N=128) -> sigmoid -> @g2
  gemm_bf16_wmma<<<dim3(2, BT_ / GBM), 256, 0, stream>>>(xg16, g1_16, hg, BT_, 128, C_);
  sig_cvt<<<(BT_ * 128 + 255) / 256, 256, 0, stream>>>(hg, hg16, BT_ * 128);
  gemm_bf16_wmma<<<gBig, 256, 0, stream>>>(hg16, g2_16, gbuf, BT_, C_, 128);

  // 7) per-head k normalization (needs a); updates k in place, writes kk
  knorm_kernel<<<BT_ * H_, 32, 0, stream>>>(kbuf, abuf, k_k, k_a, kkb);

  // 8) sequential WKV scan
  wkv_kernel<<<B_ * H_, 64, 0, stream>>>(rbuf, wlin, kbuf, vbuf, kkb, abuf, obuf);

  // 9) GroupNorm + rkv + rosa gate, emit bf16 (o*g)
  post_kernel<<<BT_, 256, 0, stream>>>(obuf, rbuf, kbuf, vbuf, gbuf, r_k,
                                       ln_w, ln_b, ids, emb, rgate, og16);

  // 10) output projection
  gemm_bf16_wmma<<<gBig, 256, 0, stream>>>(og16, Wo16, out, BT_, C_, C_);
}
